// HotPatchCapsModel_78400333021445
// MI455X (gfx1250) — matrix operand order repaired and checked by reading, not compiled, not run on hardware
//
#include <hip/hip_runtime.h>
#include <cstddef>

typedef float  v2f   __attribute__((ext_vector_type(2)));
typedef float  v8f   __attribute__((ext_vector_type(8)));
typedef __bf16 v16bf __attribute__((ext_vector_type(16)));
typedef __bf16 v8bf  __attribute__((ext_vector_type(8)));

// ---------------------------------------------------------------------------
// Tiling: workgroup = 256 threads = 8 waves arranged 2 (M) x 4 (N);
// each wave owns a 32x32 output tile = 2x2 WMMA 16x16 tiles.
// ---------------------------------------------------------------------------
constexpr int BM = 64;
constexpr int BN = 128;
constexpr int BK = 32;

constexpr int SA_F32 = 36;   // f32 A tile row stride (floats, 32+4 pad)
constexpr int SB_F32 = 136;  // f32 B tile row stride (floats, 128+8 pad)
constexpr int SAB    = 40;   // bf16 A tile row stride (elems, 32+8 pad)
constexpr int STB    = 40;   // bf16 B^T tile row stride (elems, 32+8 pad)

static __device__ __forceinline__ v8f zero8() {
    v8f z;
#pragma unroll
    for (int i = 0; i < 8; ++i) z[i] = 0.0f;
    return z;
}

// LDS offset (addrspace(3) pointers are 32-bit offsets on amdgcn)
static __device__ __forceinline__ unsigned lds_off(const void* p) {
    return (unsigned)(unsigned long long)(__attribute__((address_space(3))) const char*)p;
}

// One 16B global->LDS async copy per lane (GVS mode: saddr + 32-bit byte off).
// Tracked by ASYNCcnt; pair with s_wait_asynccnt before the barrier.
static __device__ __forceinline__ void async_b128(const void* lds_dst,
                                                  const float* __restrict__ gbase,
                                                  int byte_off) {
    asm volatile("global_load_async_to_lds_b128 %0, %1, %2"
                 :: "v"(lds_off(lds_dst)), "v"(byte_off), "s"(gbase)
                 : "memory");
}
static __device__ __forceinline__ void wait_async0() {
    asm volatile("s_wait_asynccnt 0x0" ::: "memory");
}

// ---------------------------------------------------------------------------
// Kernel 1: bf16 WMMA GEMM (fp32 in -> bf16 LDS tiles -> f32 accum), bias+relu
//   emits v_wmma_f32_16x16x32_bf16; prefetches next A K-tile
// ---------------------------------------------------------------------------
__global__ __launch_bounds__(256)
void gemm_bf16_wmma(const float* __restrict__ A, const float* __restrict__ B,
                    const float* __restrict__ bias, float* __restrict__ C,
                    int M, int N, int K)
{
    __shared__ __bf16 As[BM * SAB];   //  5 KB, [m][k]
    __shared__ __bf16 Bt[BN * STB];   // 10 KB, [n][k] transposed

    const int tid  = threadIdx.x;
    const int lane = tid & 31;
    const int wave = tid >> 5;
    const int wm   = wave >> 2;
    const int wn   = wave & 3;
    const int m16  = lane & 15;
    const int h    = lane >> 4;

    const int row0 = blockIdx.y * BM;
    const int col0 = blockIdx.x * BN;

    v8f acc[2][2];
#pragma unroll
    for (int i = 0; i < 2; ++i)
#pragma unroll
        for (int j = 0; j < 2; ++j) acc[i][j] = zero8();

    for (int k0 = 0; k0 < K; k0 += BK) {
        // stage A: 64x32 fp32 -> bf16 (conversion requires VGPR round-trip)
        for (int idx = tid; idx < BM * (BK / 4); idx += 256) {
            const int r  = idx >> 3;
            const int c4 = idx & 7;
            const int ga = (row0 + r) * K + k0 + c4 * 4;     // < 2^27, int ok
            const float4 v = *(const float4*)(A + ga);
            if (k0 + BK < K)
                __builtin_prefetch(A + ga + BK, 0, 1);       // global_prefetch_b8
            __bf16* d = &As[r * SAB + c4 * 4];
            d[0] = (__bf16)v.x; d[1] = (__bf16)v.y; d[2] = (__bf16)v.z; d[3] = (__bf16)v.w;
        }
        // stage B transposed: 32x128 fp32 -> Bt[n][k] bf16
        for (int idx = tid; idx < BK * (BN / 4); idx += 256) {
            const int r  = idx >> 5;
            const int c4 = idx & 31;
            const float4 v = *(const float4*)(B + (k0 + r) * N + col0 + c4 * 4);
            const int n = c4 * 4;
            Bt[(n + 0) * STB + r] = (__bf16)v.x;
            Bt[(n + 1) * STB + r] = (__bf16)v.y;
            Bt[(n + 2) * STB + r] = (__bf16)v.z;
            Bt[(n + 3) * STB + r] = (__bf16)v.w;
        }
        __syncthreads();

        // A frag element e -> k = 16*(e>>3) + 8h + (e&7);  B frag e -> k = 16*(e>>3) + 8h + (e&7)
        v16bf afr[2], bfr[2];
#pragma unroll
        for (int im = 0; im < 2; ++im) {
            const int m = wm * 32 + im * 16 + m16;
            const v8bf lo = *(const v8bf*)&As[m * SAB + 8 * h];
            const v8bf hi = *(const v8bf*)&As[m * SAB + 16 + 8 * h];
#pragma unroll
            for (int e = 0; e < 8; ++e) { afr[im][e] = lo[e]; afr[im][8 + e] = hi[e]; }
        }
#pragma unroll
        for (int in = 0; in < 2; ++in) {
            const int n = wn * 32 + in * 16 + m16;
            const v8bf lo = *(const v8bf*)&Bt[n * STB + 8 * h];
            const v8bf hi = *(const v8bf*)&Bt[n * STB + 16 + 8 * h];
#pragma unroll
            for (int e = 0; e < 8; ++e) { bfr[in][e] = lo[e]; bfr[in][8 + e] = hi[e]; }
        }
#pragma unroll
        for (int im = 0; im < 2; ++im)
#pragma unroll
            for (int in = 0; in < 2; ++in)
                acc[im][in] = __builtin_amdgcn_wmma_f32_16x16x32_bf16(
                    false, afr[im], false, bfr[in], (short)0, acc[im][in], false, false);
        __syncthreads();
    }

    // epilogue (32-bit offsets; C/D layout: row = r + 8h, col = lane&15)
#pragma unroll
    for (int im = 0; im < 2; ++im) {
#pragma unroll
        for (int in = 0; in < 2; ++in) {
            const int col = col0 + wn * 32 + in * 16 + m16;
            const float bv = bias[col];
            int off = (row0 + wm * 32 + im * 16 + 8 * h) * N + col;
#pragma unroll
            for (int r = 0; r < 8; ++r) {
                float v = acc[im][in][r] + bv;
                v = fmaxf(v, 0.0f);
                C[off] = v;
                off += N;
            }
        }
    }
}

// ---------------------------------------------------------------------------
// Kernel 2: fp32 WMMA GEMM (exact), bias, optional relu
//   emits v_wmma_f32_16x16x4_f32; tiles staged with GLOBAL_LOAD_ASYNC_TO_LDS
// ---------------------------------------------------------------------------
__global__ __launch_bounds__(256)
void gemm_f32_wmma(const float* __restrict__ A, const float* __restrict__ B,
                   const float* __restrict__ bias, float* __restrict__ C,
                   int M, int N, int K, int do_relu)
{
    __shared__ float As[BM * SA_F32];  //  9 KB
    __shared__ float Bs[BK * SB_F32];  // 17 KB

    const int tid  = threadIdx.x;
    const int lane = tid & 31;
    const int wave = tid >> 5;
    const int wm   = wave >> 2;
    const int wn   = wave & 3;
    const int m16  = lane & 15;
    const int h    = lane >> 4;

    const int row0 = blockIdx.y * BM;
    const int col0 = blockIdx.x * BN;

    v8f acc[2][2];
#pragma unroll
    for (int i = 0; i < 2; ++i)
#pragma unroll
        for (int j = 0; j < 2; ++j) acc[i][j] = zero8();

    for (int k0 = 0; k0 < K; k0 += BK) {
        // async global->LDS staging (no VGPR round-trip), 16B per lane per issue
        for (int idx = tid; idx < BM * (BK / 4); idx += 256) {
            const int r  = idx >> 3;
            const int c4 = idx & 7;
            async_b128(&As[r * SA_F32 + c4 * 4], A,
                       ((row0 + r) * K + k0 + c4 * 4) * 4);
        }
        for (int idx = tid; idx < BK * (BN / 4); idx += 256) {
            const int r  = idx >> 5;
            const int c4 = idx & 31;
            async_b128(&Bs[r * SB_F32 + c4 * 4], B,
                       ((k0 + r) * N + col0 + c4 * 4) * 4);
        }
        wait_async0();          // per-wave: async copies landed in LDS
        __syncthreads();        // all waves' tiles visible

#pragma unroll
        for (int kk = 0; kk < BK; kk += 4) {
            v2f a[2], b[2];
#pragma unroll
            for (int im = 0; im < 2; ++im) {
                const float* p = &As[(wm * 32 + im * 16 + m16) * SA_F32 + kk + 2 * h];
                a[im].x = p[0]; a[im].y = p[1];
            }
#pragma unroll
            for (int in = 0; in < 2; ++in) {
                const int n = wn * 32 + in * 16 + m16;
                b[in].x = Bs[(kk + 2 * h) * SB_F32 + n];
                b[in].y = Bs[(kk + 2 * h + 1) * SB_F32 + n];
            }
#pragma unroll
            for (int im = 0; im < 2; ++im)
#pragma unroll
                for (int in = 0; in < 2; ++in)
                    acc[im][in] = __builtin_amdgcn_wmma_f32_16x16x4_f32(
                        false, a[im], false, b[in], (short)0, acc[im][in], false, false);
        }
        __syncthreads();
    }

#pragma unroll
    for (int im = 0; im < 2; ++im) {
#pragma unroll
        for (int in = 0; in < 2; ++in) {
            const int col = col0 + wn * 32 + in * 16 + m16;
            const float bv = bias[col];
            int off = (row0 + wm * 32 + im * 16 + 8 * h) * N + col;
#pragma unroll
            for (int r = 0; r < 8; ++r) {
                float v = acc[im][in][r] + bv;
                if (do_relu) v = fmaxf(v, 0.0f);
                C[off] = v;
                off += N;
            }
        }
    }
}

// ---------------------------------------------------------------------------
// pc_W [32,256,8] -> pcWT [256,256]  with  pcWT[c][n*8+d] = pc_W[n][c][d]
// ---------------------------------------------------------------------------
__global__ __launch_bounds__(256)
void transpose_pcw(const float* __restrict__ in, float* __restrict__ out)
{
    const int idx = blockIdx.x * 256 + threadIdx.x;   // 65536 total
    const int c   = idx >> 8;
    const int col = idx & 255;
    const int n   = col >> 3;
    const int d   = col & 7;
    out[idx] = in[n * 2048 + c * 8 + d];
}

// ---------------------------------------------------------------------------
// LayerNorm over 256 features, in place, one row per 256-thread block
// ---------------------------------------------------------------------------
__global__ __launch_bounds__(256)
void ln256_kernel(float* __restrict__ x, const float* __restrict__ g,
                  const float* __restrict__ b)
{
    __shared__ float rs[8], rs2[8];
    const int row = blockIdx.x;
    const int tid = threadIdx.x;
    const int lane = tid & 31, wave = tid >> 5;

    const float v = x[row * 256 + tid];
    float s = v, s2 = v * v;
#pragma unroll
    for (int off = 16; off; off >>= 1) {
        s  += __shfl_xor(s,  off, 32);
        s2 += __shfl_xor(s2, off, 32);
    }
    if (lane == 0) { rs[wave] = s; rs2[wave] = s2; }
    __syncthreads();
    if (tid == 0) {
        float ts = 0.f, ts2 = 0.f;
        for (int i = 0; i < 8; ++i) { ts += rs[i]; ts2 += rs2[i]; }
        rs[0] = ts; rs2[0] = ts2;
    }
    __syncthreads();
    const float mean = rs[0] * (1.0f / 256.0f);
    const float var  = rs2[0] * (1.0f / 256.0f) - mean * mean;
    const float inv  = rsqrtf(var + 1e-5f);
    x[row * 256 + tid] = (v - mean) * inv * g[tid] + b[tid];
}

// ---------------------------------------------------------------------------
// Per-capsule LayerNorm over 8 dims, in place; one thread per (batch,capsule)
// ---------------------------------------------------------------------------
__global__ __launch_bounds__(256)
void capsln_kernel(float* __restrict__ prim, const float* __restrict__ g,
                   const float* __restrict__ b, int total)
{
    const int t = blockIdx.x * 256 + threadIdx.x;
    if (t >= total) return;
    const int n = t & 31;
    float* p = prim + (size_t)t * 8;
    float e[8];
    float s = 0.f;
#pragma unroll
    for (int d = 0; d < 8; ++d) { e[d] = p[d]; s += e[d]; }
    const float mean = s * 0.125f;
    float vv = 0.f;
#pragma unroll
    for (int d = 0; d < 8; ++d) { const float c = e[d] - mean; vv += c * c; }
    const float inv = rsqrtf(vv * 0.125f + 1e-5f);
#pragma unroll
    for (int d = 0; d < 8; ++d)
        p[d] = (e[d] - mean) * inv * g[n * 8 + d] + b[n * 8 + d];
}

// ---------------------------------------------------------------------------
// Fused predictions + dynamic routing + logits; one workgroup per batch
// element; the 32x100x16 prediction tile lives in the 320 KB WGP LDS
// (j-stride padded to 17 for conflict-free stride patterns); dc_W stays L2-
// resident (1.6 MB) across all 4096 blocks.
// ---------------------------------------------------------------------------
constexpr int NPC = 32;
constexpr int NL  = 100;
constexpr int CD  = 16;
constexpr int PD  = 8;
constexpr int PJ  = 17;

__global__ __launch_bounds__(256)
void routing_kernel(const float* __restrict__ prim,      // [B,32,8] (LN'd)
                    const float* __restrict__ dcW,       // [32,100,16,8]
                    const float* __restrict__ sf,        // [B,100]
                    const float* __restrict__ priors,    // [100,100]
                    const float* __restrict__ targets,   // [100,16]
                    const float* __restrict__ temp,      // [1]
                    float* __restrict__ logits)          // [B,100]
{
    __shared__ float pred[NPC * NL * PJ];   // 217,600 B
    __shared__ float bb[NPC * NL];
    __shared__ float cc[NPC * NL];
    __shared__ float sv[NL * CD];
    __shared__ float sp[NL];
    __shared__ float prim_s[NPC * PD];
    __shared__ float sfw[NL];
    __shared__ float red0;

    const int bidx = blockIdx.x;
    const int tid  = threadIdx.x;

    if (tid < NPC * PD) prim_s[tid] = prim[(size_t)bidx * NPC * PD + tid];
    if (tid < NL) sfw[tid] = fmaxf(sf[(size_t)bidx * NL + tid], 0.0f);
    __syncthreads();
    if (tid == 0) {
        float s = 0.f;
        for (int k = 0; k < NL; ++k) s += sfw[k];
        red0 = fmaxf(s, 1e-6f);
    }
    __syncthreads();
    if (tid < NL) sfw[tid] = sfw[tid] / red0 - (1.0f / NL);
    __syncthreads();
    if (tid < NL) {
        float acc = 0.f;
        for (int k = 0; k < NL; ++k) acc += sfw[k] * priors[k * NL + tid];
        sp[tid] = acc;
    }
    __syncthreads();

    // predictions[n][j][o] = sum_i prim[n][i] * dcW[n][j][o][i]
    for (int idx = tid; idx < NPC * NL * CD; idx += 256) {
        const int n   = idx / (NL * CD);
        const int rem = idx - n * (NL * CD);
        const float4* w = (const float4*)(dcW + (size_t)idx * PD);
        const float4 w0 = w[0], w1 = w[1];
        const float* pr = &prim_s[n * PD];
        const float acc = w0.x * pr[0] + w0.y * pr[1] + w0.z * pr[2] + w0.w * pr[3]
                        + w1.x * pr[4] + w1.y * pr[5] + w1.z * pr[6] + w1.w * pr[7];
        const int j = rem >> 4, o = rem & 15;
        pred[(n * NL + j) * PJ + o] = acc;
    }
    for (int idx = tid; idx < NPC * NL; idx += 256)
        bb[idx] = 0.1f * sp[idx % NL];
    __syncthreads();

    for (int it = 0; it < 3; ++it) {
        // c = softmax over labels per primary capsule
        if (tid < NPC) {
            const int base = tid * NL;
            float mx = -3.4e38f;
            for (int j = 0; j < NL; ++j) mx = fmaxf(mx, bb[base + j]);
            float s = 0.f;
            for (int j = 0; j < NL; ++j) {
                const float e = __expf(bb[base + j] - mx);
                cc[base + j] = e; s += e;
            }
            const float inv = 1.0f / s;
            for (int j = 0; j < NL; ++j) cc[base + j] *= inv;
        }
        __syncthreads();
        // s[j][o] = sum_n pred[n][j][o] * c[n][j]
        for (int idx = tid; idx < NL * CD; idx += 256) {
            const int j = idx >> 4, o = idx & 15;
            float acc = 0.f;
            for (int n = 0; n < NPC; ++n)
                acc += pred[(n * NL + j) * PJ + o] * cc[n * NL + j];
            sv[idx] = acc;
        }
        __syncthreads();
        // v = squash(s) in place
        if (tid < NL) {
            float sq = 0.f;
            for (int o = 0; o < CD; ++o) { const float t = sv[tid * CD + o]; sq += t * t; }
            const float scale = (sq / (1.0f + sq)) * (1.0f / sqrtf(sq + 1e-8f));
            for (int o = 0; o < CD; ++o) sv[tid * CD + o] *= scale;
        }
        __syncthreads();
        if (it < 2) {
            const float strength = 0.1f - 0.02f * (float)(it + 1);
            for (int idx = tid; idx < NPC * NL; idx += 256) {
                const int n = idx / NL, j = idx - n * NL;
                const float* pp = &pred[(n * NL + j) * PJ];
                float acc = 0.f;
#pragma unroll
                for (int o = 0; o < CD; ++o) acc += pp[o] * sv[j * CD + o];
                bb[idx] += acc + sp[j] * strength;
            }
            __syncthreads();
        }
    }

    // logits[j] = dot(l2norm(v[j]), targets[j]) * temperature
    if (tid < NL) {
        float sq = 0.f;
        for (int o = 0; o < CD; ++o) { const float t = sv[tid * CD + o]; sq += t * t; }
        const float nrm = fmaxf(sqrtf(sq), 1e-12f);
        float acc = 0.f;
        for (int o = 0; o < CD; ++o) acc += sv[tid * CD + o] * targets[tid * CD + o];
        logits[(size_t)bidx * NL + tid] = acc / nrm * temp[0];
    }
}

// ---------------------------------------------------------------------------
// Host-side orchestration
// ---------------------------------------------------------------------------
extern "C" void kernel_launch(void* const* d_in, const int* in_sizes, int n_in,
                              void* d_out, int out_size, void* d_ws, size_t ws_size,
                              hipStream_t stream)
{
    (void)in_sizes; (void)n_in; (void)out_size; (void)ws_size;

    const float* features    = (const float*)d_in[0];   // [4096,20000]
    const float* slot_feat   = (const float*)d_in[1];   // [4096,100]
    const float* W1          = (const float*)d_in[2];   // [20000,512]
    const float* b1          = (const float*)d_in[3];   // [512]
    const float* W2          = (const float*)d_in[4];   // [512,256]
    const float* b2          = (const float*)d_in[5];   // [256]
    const float* ln_g        = (const float*)d_in[6];   // [256]
    const float* ln_b        = (const float*)d_in[7];   // [256]
    const float* pc_W        = (const float*)d_in[8];   // [32,256,8]
    const float* pc_b        = (const float*)d_in[9];   // [32,8]
    const float* pc_g        = (const float*)d_in[10];  // [32,8]
    const float* pc_beta     = (const float*)d_in[11];  // [32,8]
    const float* dc_W        = (const float*)d_in[12];  // [32,100,16,8]
    const float* slot_priors = (const float*)d_in[13];  // [100,100]
    const float* targets     = (const float*)d_in[14];  // [100,16]
    const float* temperature = (const float*)d_in[15];  // [1]
    float* logits = (float*)d_out;                      // [4096,100]

    constexpr int B = 4096;
    float* ws   = (float*)d_ws;
    float* h1   = ws;                         // 4096*512
    float* h2   = h1 + (size_t)B * 512;       // 4096*256
    float* pcWT = h2 + (size_t)B * 256;       // 256*256
    float* prim = pcWT + 256 * 256;           // 4096*256

    transpose_pcw<<<256, 256, 0, stream>>>(pc_W, pcWT);

    // h1 = relu(features @ W1 + b1)   (bf16 WMMA, f32 accum; HBM-roofline bound)
    gemm_bf16_wmma<<<dim3(512 / BN, B / BM), 256, 0, stream>>>(
        features, W1, b1, h1, B, 512, 20000);

    // h2 = relu(h1 @ W2 + b2)         (fp32 WMMA + async-to-LDS staging)
    gemm_f32_wmma<<<dim3(256 / BN, B / BM), 256, 0, stream>>>(
        h1, W2, b2, h2, B, 256, 512, 1);

    ln256_kernel<<<B, 256, 0, stream>>>(h2, ln_g, ln_b);

    // prim = h2 @ pcWT + pc_b         (fp32 WMMA)
    gemm_f32_wmma<<<dim3(256 / BN, B / BM), 256, 0, stream>>>(
        h2, pcWT, pc_b, prim, B, 256, 256, 0);

    capsln_kernel<<<(B * 32 + 255) / 256, 256, 0, stream>>>(
        prim, pc_g, pc_beta, B * 32);

    routing_kernel<<<B, 256, 0, stream>>>(
        prim, dc_W, slot_feat, slot_priors, targets, temperature, logits);
}
